// equi_conv8_21603685499537
// MI455X (gfx1250) — matrix operand order repaired and checked by reading, not compiled, not run on hardware
//
#include <hip/hip_runtime.h>
#include <hip/hip_bf16.h>

typedef __attribute__((ext_vector_type(16))) _Float16 v16h;
typedef __attribute__((ext_vector_type(8)))  float    v8f;

// ---------------- constant tables (precomputed from the reference) ----------
#define ICO_A 0.52573111f
#define ICO_B 0.85065081f

__constant__ int d_INV[12][6] = {
  {0,1,2,3,4,5},{0,5,4,3,2,1},{4,3,0,5,2,1},{1,2,4,3,5,0},
  {3,5,2,0,4,1},{1,4,3,5,0,2},{4,0,3,1,2,5},{1,0,2,4,3,5},
  {4,1,2,5,0,3},{3,1,4,0,2,5},{2,1,4,5,3,0},{4,5,0,3,1,2}};
__constant__ int d_ROLL[5][6] = {
  {0,1,2,3,4,5},{0,2,3,4,5,1},{0,3,4,5,1,2},{0,4,5,1,2,3},{0,5,1,2,3,4}};
__constant__ int d_C2V[6][2] = {{0,1},{6,7},{2,11},{4,9},{5,8},{3,10}};
__constant__ float d_VS[12][3] = {
  {0.f, ICO_A, ICO_B},{0.f, ICO_A,-ICO_B},{0.f,-ICO_A, ICO_B},{0.f,-ICO_A,-ICO_B},
  { ICO_A, ICO_B,0.f},{ ICO_A,-ICO_B,0.f},{-ICO_A, ICO_B,0.f},{-ICO_A,-ICO_B,0.f},
  { ICO_B,0.f, ICO_A},{ ICO_B,0.f,-ICO_A},{-ICO_B,0.f, ICO_A},{-ICO_B,0.f,-ICO_A}};

// problem constants
#define Vn    2048
#define NNB   16
#define DI    32
#define DO    64
#define CI    192      // DI*6 contraction dim
#define KSTEPS 6       // 192 / 32

// _build_sym source-parameter index for (k13, color)
__device__ __forceinline__ int src_idx(int k13, int col) {
  if (k13 == 0)  return (col == 0) ? 0 : 1;
  if (k13 == 1)  return (col == 0) ? 2 : 3;
  if (k13 == 12) return (col == 0) ? 4 : 5;
  int j = k13 - 2; int s = j / 5, q = j - 5 * s;
  return 6 + s * 6 + d_ROLL[q][col];
}

// A-tile mapping (16-bit A 16x32, contiguous k, same order as B): element(m,kk) -> lane,slot
//   lane = m + 16*(kk>>4) ; slot = kk&15
// inverse used below: kk = slot | (((lane>>4)&1)<<4)
// B-tile mapping (32x16): lane = n + 16*(kk>>4) ; slot = kk&15

// ---------------- K0a: expand W_dir -> Aact tiles [k12][mt6][ks6][512] f16 --
__global__ __launch_bounds__(256) void build_Aact(const float* __restrict__ Wdir,
                                                  _Float16* __restrict__ Aact) {
  int idx = blockIdx.x * 256 + threadIdx.x;
  if (idx >= 12 * 6 * 6 * 512) return;
  int e = idx & 511, t = idx >> 9;
  int lane = e >> 4, slot = e & 15;
  int ks = t % 6; int mt = (t / 6) % 6; int k = t / 36;
  int m  = lane & 15;
  int kk = slot | (((lane >> 4) & 1) << 4);
  int row = mt * 16 + m;            // (h,r): row = h*12 + r, 0..95
  int h = row / 12, r = row - 12 * h;
  int ci = ks * 32 + kk;            // c*6 + i
  int c = ci / 6, i = ci - 6 * c;
  int col = d_INV[r][i];
  float v = 0.5f * Wdir[(h * 32 + c) * 18 + src_idx(k, col)];  // 1/sqrt(dph)=0.5 folded
  Aact[idx] = (_Float16)v;
}

// ---------------- K0b: expand W -> Afeat tiles [k13][r12][mt4][ks6][512] ----
__global__ __launch_bounds__(256) void build_Afeat(const float* __restrict__ W,
                                                   _Float16* __restrict__ Afeat) {
  int idx = blockIdx.x * 256 + threadIdx.x;
  if (idx >= 13 * 12 * 4 * 6 * 512) return;
  int e = idx & 511, t = idx >> 9;
  int lane = e >> 4, slot = e & 15;
  int ks = t % 6; int mt = (t / 6) % 4; int r = (t / 24) % 12; int k13 = t / 288;
  int m  = lane & 15;
  int kk = slot | (((lane >> 4) & 1) << 4);
  int d = mt * 16 + m;              // 0..63
  int ci = ks * 32 + kk;
  int c = ci / 6, i = ci - 6 * c;
  int col = d_INV[r][i];
  Afeat[idx] = (_Float16)W[(d * 32 + c) * 18 + src_idx(k13, col)];
}

// ---------------- K1: cross (W_dim applied, direction feats) -> B tiles -----
// cross16 layout: [p][ks6][512] f16, B-tile mapping with n = column.
__global__ __launch_bounds__(256) void prep_cross(const int* __restrict__ nbr,
                                                  const float* __restrict__ verts,
                                                  const float* __restrict__ fm,
                                                  const float* __restrict__ Wdim,
                                                  _Float16* __restrict__ cross16) {
  __shared__ float nfL[DI * NNB * 6];   // [c][n][i]
  __shared__ float de6L[NNB][6];
  __shared__ float wL[33 * 32];
  __shared__ int   idxL[NNB];
  int tid = threadIdx.x, p = blockIdx.x;

  if (tid < NNB) {
    int gi = nbr[p * NNB + tid];
    idxL[tid] = gi;
    float cx = verts[p * 3], cy = verts[p * 3 + 1], cz = verts[p * 3 + 2];
    float dx = verts[gi * 3]     - cx;
    float dy = verts[gi * 3 + 1] - cy;
    float dz = verts[gi * 3 + 2] - cz;
    float nrm = fmaxf(sqrtf(dx * dx + dy * dy + dz * dz), 1e-12f);
    dx /= nrm; dy /= nrm; dz /= nrm;
    float de12[12];
    #pragma unroll
    for (int v = 0; v < 12; ++v)
      de12[v] = d_VS[v][0] * dx + d_VS[v][1] * dy + d_VS[v][2] * dz;
    #pragma unroll
    for (int j = 0; j < 6; ++j)
      de6L[tid][j] = fmaxf(de12[d_C2V[j][0]], de12[d_C2V[j][1]]);
  }
  for (int t = tid; t < 33 * 32; t += 256) wL[t] = Wdim[t];
  __syncthreads();
  for (int t = tid; t < DI * NNB * 6; t += 256) {
    int c = t / 96, rem = t - 96 * c, n = rem / 6, i = rem - 6 * n;
    nfL[t] = fm[(c * Vn + idxL[n]) * 6 + i];
  }
  __syncthreads();
  for (int t = tid; t < NNB * CI; t += 256) {     // 3072 outputs
    int n = t / CI, ci = t - CI * n;
    int o = ci / 6, i = ci - 6 * o;
    float sum = wL[32 * 32 + o] * de6L[n][i];     // channel 32 = direction feature
    #pragma unroll
    for (int c = 0; c < DI; ++c) sum += wL[c * 32 + o] * nfL[(c * NNB + n) * 6 + i];
    int ks = ci >> 5, kk = ci & 31;
    cross16[(p * KSTEPS + ks) * 512 + (n + ((kk >> 4) << 4)) * 16 + (kk & 15)] = (_Float16)sum;
  }
}

// ---------------- K1b: feature_map as B tiles per 16-vertex block -----------
__global__ __launch_bounds__(256) void prep_fm16(const float* __restrict__ fm,
                                                 _Float16* __restrict__ fm16) {
  int b = blockIdx.x, p0 = b * 16;
  for (int t = threadIdx.x; t < KSTEPS * 512; t += 256) {
    int ks = t / 512, e = t - 512 * ks;
    int lane = e >> 4, slot = e & 15;
    int pl = lane & 15;
    int kk = slot | ((lane >> 4) << 4);
    int ci = ks * 32 + kk; int c = ci / 6, i = ci - 6 * c;
    fm16[(b * KSTEPS + ks) * 512 + e] = (_Float16)fm[(c * Vn + p0 + pl) * 6 + i];
  }
}

// ---------------- K2: fused attention + symmetry conv (WMMA) ----------------
// LDS: crossL 96KB | actL 48KB | nfmL 12KB = 156KB (kaL overlays crossL at end)
__global__ __launch_bounds__(256) void fused_attn(const _Float16* __restrict__ Aact,
                                                  const _Float16* __restrict__ Afeat,
                                                  const _Float16* __restrict__ cross16,
                                                  const _Float16* __restrict__ fm16,
                                                  float* __restrict__ out) {
  __shared__ __align__(64) char smem[159744];
  _Float16* crossL = (_Float16*)smem;             // [p16][ks6][512] f16 = 96KB
  _Float16* actL   = (_Float16*)(smem + 98304);   // [p16][h8][r12][n16] f16 = 48KB
  _Float16* nfmL   = (_Float16*)(smem + 147456);  // [r12][512] B tiles = 12KB
  float*    kaL    = (float*)smem;                // overlay: [d64][p16][r12] f32 = 48KB

  const int tid  = threadIdx.x;
  const int w    = tid >> 5;
  const int lane = tid & 31;
  const int b    = blockIdx.x;
  const int p0   = b * 16;

  // ---- async global->LDS copy of this block's contiguous cross slice (96KB)
  {
    const _Float16* src = cross16 + (size_t)p0 * KSTEPS * 512;
    unsigned ldsbase = (unsigned)(uintptr_t)crossL;     // addrspace(3) offset in low 32b
    unsigned long long gbase = (unsigned long long)(uintptr_t)src;
    #pragma unroll
    for (int it = 0; it < 24; ++it) {                   // 24 * 256 lanes * 16B = 96KB
      unsigned loff = ldsbase + it * 4096 + tid * 16;
      unsigned long long g = gbase + it * 4096 + tid * 16;
      asm volatile("global_load_async_to_lds_b128 %0, %1, off"
                   :: "v"(loff), "v"(g) : "memory");
    }
    asm volatile("s_wait_asynccnt 0x0" ::: "memory");
  }
  __syncthreads();

  v8f accf[6];
  #pragma unroll
  for (int t = 0; t < 6; ++t) accf[t] = (v8f){0.f,0.f,0.f,0.f,0.f,0.f,0.f,0.f};

  for (int k = 0; k < 12; ++k) {
    __syncthreads();   // actL reusable
    // ---- GEMM1: logits (96 rows (h,r)) x (16 cols = neighbors of one vertex)
    // mt-outer: keep the 6 A-tiles (48 VGPRs) resident for both vertices of this wave
    for (int mt = 0; mt < 6; ++mt) {
      v16h a[KSTEPS];
      #pragma unroll
      for (int ks = 0; ks < KSTEPS; ++ks)
        a[ks] = *(const v16h*)(Aact + (((k * 6 + mt) * 6) + ks) * 512 + lane * 16);
      #pragma unroll
      for (int pp = 0; pp < 2; ++pp) {
        int p = w + 8 * pp;                 // wave w owns vertices w and w+8
        v8f acc = (v8f){0.f,0.f,0.f,0.f,0.f,0.f,0.f,0.f};
        #pragma unroll
        for (int ks = 0; ks < KSTEPS; ++ks) {
          v16h bb = *(const v16h*)(crossL + (p * KSTEPS + ks) * 512 + lane * 16);
          acc = __builtin_amdgcn_wmma_f32_16x16x32_f16(false, a[ks], false, bb,
                                                       (short)0, acc, false, false);
        }
        // softmax over the 16 columns (n) of each row: half-wave reductions
        #pragma unroll
        for (int v = 0; v < 8; ++v) {
          float val = acc[v];
          float mx = val;
          mx = fmaxf(mx, __shfl_xor(mx, 1, 16));
          mx = fmaxf(mx, __shfl_xor(mx, 2, 16));
          mx = fmaxf(mx, __shfl_xor(mx, 4, 16));
          mx = fmaxf(mx, __shfl_xor(mx, 8, 16));
          float e = __expf(val - mx);
          float s = e;
          s += __shfl_xor(s, 1, 16);
          s += __shfl_xor(s, 2, 16);
          s += __shfl_xor(s, 4, 16);
          s += __shfl_xor(s, 8, 16);
          float aout = e / s;
          int row = mt * 16 + v + 8 * (lane >> 4);
          int h = row / 12, r = row - 12 * h, n = lane & 15;
          actL[((p * 8 + h) * 12 + r) * 16 + n] = (_Float16)aout;
        }
      }
    }
    __syncthreads();
    // hint: pull next k's Aact block toward the caches (global_prefetch_b8)
    if (k < 11) __builtin_prefetch(Aact + ((k + 1) * 36) * 512, 0, 2);
    // ---- per K-step: build nfm B tiles (VALU from LDS), then GEMM2 --------
    for (int ks = 0; ks < KSTEPS; ++ks) {
      #pragma unroll
      for (int pair = 0; pair < 2; ++pair) {
        int pk = tid + 256 * pair;       // 0..511 -> (p,kk)
        int p = pk >> 5, kk = pk & 31;
        int ci = ks * 32 + kk;
        int c = ci / 6, i = ci - 6 * c, h = c >> 2;
        int hi = (kk >> 4) << 4, sl = kk & 15;
        const _Float16* cr = crossL + (p * KSTEPS + ks) * 512;
        float cv[16];
        #pragma unroll
        for (int n = 0; n < 16; ++n) cv[n] = (float)cr[(n + hi) * 16 + sl];
        for (int r = 0; r < 12; ++r) {
          const _Float16* ar = actL + ((p * 8 + h) * 12 + r) * 16;
          float sum = 0.f;
          #pragma unroll
          for (int n = 0; n < 16; ++n) sum += (float)ar[n] * cv[n];
          nfmL[r * 512 + (p + hi) * 16 + sl] = (_Float16)sum;
        }
      }
      __syncthreads();
      #pragma unroll
      for (int t = 0; t < 6; ++t) {
        int j2 = w + 8 * t;          // 0..47
        int r = j2 % 12, mt = j2 / 12;
        v16h a  = *(const v16h*)(Afeat + (((k * 12 + r) * 4 + mt) * 6 + ks) * 512 + lane * 16);
        v16h bb = *(const v16h*)(nfmL + r * 512 + lane * 16);
        accf[t] = __builtin_amdgcn_wmma_f32_16x16x32_f16(false, a, false, bb,
                                                         (short)0, accf[t], false, false);
      }
      __syncthreads();
    }
  }
  // ---- center term: k13 = 12, B = feature_map tiles --------------------------
  for (int ks = 0; ks < KSTEPS; ++ks) {
    v16h bb = *(const v16h*)(fm16 + (b * KSTEPS + ks) * 512 + lane * 16);
    #pragma unroll
    for (int t = 0; t < 6; ++t) {
      int j2 = w + 8 * t;
      int r = j2 % 12, mt = j2 / 12;
      v16h a = *(const v16h*)(Afeat + (((12 * 12 + r) * 4 + mt) * 6 + ks) * 512 + lane * 16);
      accf[t] = __builtin_amdgcn_wmma_f32_16x16x32_f16(false, a, false, bb,
                                                       (short)0, accf[t], false, false);
    }
  }
  __syncthreads();   // done reading crossL/actL/nfmL -> overlay kaL on crossL
  #pragma unroll
  for (int t = 0; t < 6; ++t) {
    int j2 = w + 8 * t;
    int r = j2 % 12, mt = j2 / 12;
    #pragma unroll
    for (int v = 0; v < 8; ++v) {
      int d = mt * 16 + v + 8 * (lane >> 4);
      int p = lane & 15;
      kaL[(d * 16 + p) * 12 + r] = accf[t][v];
    }
  }
  __syncthreads();
  // ---- final max over color-to-vertex pairs, write ka (do, V, 6) -------------
  #pragma unroll
  for (int it = 0; it < 24; ++it) {
    int idx = tid + 256 * it;          // < 6144
    int d = idx / 96, rem = idx - 96 * d;
    int p = rem / 6, j = rem - 6 * p;
    float a0 = kaL[(d * 16 + p) * 12 + d_C2V[j][0]];
    float a1 = kaL[(d * 16 + p) * 12 + d_C2V[j][1]];
    out[(d * Vn + (p0 + p)) * 6 + j] = fmaxf(a0, a1);
  }
}

// ---------------- host launch ------------------------------------------------
extern "C" void kernel_launch(void* const* d_in, const int* in_sizes, int n_in,
                              void* d_out, int out_size, void* d_ws, size_t ws_size,
                              hipStream_t stream) {
  (void)in_sizes; (void)n_in; (void)out_size; (void)ws_size;
  const int*   nbr   = (const int*)d_in[0];
  const float* verts = (const float*)d_in[1];
  const float* fm    = (const float*)d_in[2];
  const float* Wdim  = (const float*)d_in[3];
  const float* W     = (const float*)d_in[4];
  const float* Wdir  = (const float*)d_in[5];
  float* out = (float*)d_out;

  char* ws = (char*)d_ws;
  // workspace layout (bytes, all 32B aligned)
  _Float16* Aact    = (_Float16*)(ws);                         // 12*6*6*512   f16 =   442,368 B
  _Float16* Afeat   = (_Float16*)(ws + 442368);                // 13*12*4*6*512 f16 = 3,833,856 B
  _Float16* cross16 = (_Float16*)(ws + 4276224);               // 2048*6*512 f16    = 12,582,912 B
  _Float16* fm16    = (_Float16*)(ws + 16859136);              // 128*6*512 f16     =   786,432 B

  build_Aact <<<(12 * 6 * 6 * 512 + 255) / 256, 256, 0, stream>>>(Wdir, Aact);
  build_Afeat<<<(13 * 12 * 4 * 6 * 512 + 255) / 256, 256, 0, stream>>>(W, Afeat);
  prep_cross <<<Vn, 256, 0, stream>>>(nbr, verts, fm, Wdim, cross16);
  prep_fm16  <<<Vn / 16, 256, 0, stream>>>(fm, fm16);
  fused_attn <<<Vn / 16, 256, 0, stream>>>(Aact, Afeat, cross16, fm16, out);

  // second tuple output: feature_map passthrough (32*2048*6 floats after ka)
  hipMemcpyAsync(out + (size_t)DO * Vn * 6, fm,
                 (size_t)DI * Vn * 6 * sizeof(float),
                 hipMemcpyDeviceToDevice, stream);
}
